// DeltaNet_14637248545326
// MI455X (gfx1250) — matrix operand order repaired and checked by reading, not compiled, not run on hardware
//
#include <hip/hip_runtime.h>
#include <hip/hip_bf16.h>

// ---------------------------------------------------------------------------
// Gated linear attention (DeltaNet-style) layer for MI455X (gfx1250).
// All dense math on v_wmma_f32_16x16x32_bf16 (fp32 accum); recurrence is
// chunk-parallelized (C=64) with an fp32 state scan.
// ---------------------------------------------------------------------------

typedef __attribute__((ext_vector_type(16))) __bf16 v16bf;
typedef __attribute__((ext_vector_type(8)))  float  v8f;

#define DEV __device__ __forceinline__

constexpr int kB   = 4;
constexpr int kS   = 4096;
constexpr int kDM  = 1024;
constexpr int kH   = 16;
constexpr int kDH  = 64;
constexpr int kC   = 64;            // chunk length
constexpr int kNC  = kS / kC;       // 64 chunks per sequence
constexpr int kROWS = kB * kS;      // 16384 token rows

DEV v8f zero8() {
  v8f z;
#pragma unroll
  for (int i = 0; i < 8; ++i) z[i] = 0.f;
  return z;
}

DEV v8f wmma_bf16(v16bf a, v16bf b, v8f c) {
  // D = A(16x32 bf16) * B(32x16 bf16) + C(16x16 f32)
  return __builtin_amdgcn_wmma_f32_16x16x32_bf16(
      /*neg_a=*/false, a, /*neg_b=*/false, b,
      /*c_mod=*/(short)0, c, /*reuse_a=*/false, /*reuse_b=*/false);
}

DEV float fast_rcp(float x) { return __builtin_amdgcn_rcpf(x); }  // v_rcp_f32

// ---- fragment loaders: K-contiguous sources ----
// A fragment: lane holds row (L%16); K = 16 consecutive from kb=(L>=16)?16:0
DEV v16bf a_frag_f32(const float* row, int lanehi) {
  const int kb = lanehi ? 16 : 0;
  v16bf f;
#pragma unroll
  for (int i = 0; i < 8; ++i) f[i] = (__bf16)row[kb + i];
#pragma unroll
  for (int i = 0; i < 8; ++i) f[8 + i] = (__bf16)row[kb + 8 + i];
  return f;
}
DEV v16bf a_frag_bf(const __bf16* row, int lanehi) {
  const int kb = lanehi ? 16 : 0;
  v16bf f;
#pragma unroll
  for (int i = 0; i < 8; ++i) f[i] = row[kb + i];
#pragma unroll
  for (int i = 0; i < 8; ++i) f[8 + i] = row[kb + 8 + i];
  return f;
}
// B fragment: lane holds column (L%16); K = 16 consecutive from kb=(L>=16)?16:0
DEV v16bf b_frag_f32(const float* row, int lanehi) {
  const int kb = lanehi ? 16 : 0;
  v16bf f;
#pragma unroll
  for (int i = 0; i < 16; ++i) f[i] = (__bf16)row[kb + i];
  return f;
}
DEV v16bf b_frag_bf(const __bf16* row, int lanehi) {
  const int kb = lanehi ? 16 : 0;
  v16bf f;
#pragma unroll
  for (int i = 0; i < 16; ++i) f[i] = row[kb + i];
  return f;
}

DEV float phi_elu(float v) {               // elu(v)+1+1e-6
  return v > 0.f ? v + 1.000001f : __expf(v) + 1e-6f;
}
DEV float head_logr(const float* theta, int h) {
  // r = exp(-1/(softplus(theta)+1));  return log r = -1/(softplus+1)
  float t  = theta[h];
  float sp = (t > 20.f) ? t : log1pf(__expf(t));
  return -1.0f / (sp + 1.0f);
}

// ---------------------------------------------------------------------------
// Kernel: gate = sigmoid(x @ Wg^T + bg) * mask        (ROWS x 16, WMMA N=16)
// ---------------------------------------------------------------------------
__global__ __launch_bounds__(32)
void gate_kernel(const float* __restrict__ X, const float* __restrict__ Wg,
                 const float* __restrict__ bg, const unsigned char* __restrict__ mask,
                 float* __restrict__ gate) {
  const int L = threadIdx.x;
  const int lr = L & 15, lanehi = (L >> 4) & 1;
  const int r0 = blockIdx.x * 16;
  const float* arow = X + (size_t)(r0 + lr) * kDM;
  const float* brow = Wg + (size_t)lr * kDM;          // Wg is (H=16, DM) row-major
  v8f acc = zero8();
  for (int k0 = 0; k0 < kDM; k0 += 32) {
    v16bf a = a_frag_f32(arow + k0, lanehi);
    v16bf b = b_frag_f32(brow + k0, lanehi);
    acc = wmma_bf16(a, b, acc);
  }
#pragma unroll
  for (int i = 0; i < 8; ++i) {
    int row = r0 + i + 8 * lanehi;
    float m = mask[row] ? 1.f : 0.f;
    float g = acc[i] + bg[lr];
    g = fast_rcp(1.f + __expf(-g));                   // fast sigmoid
    gate[(size_t)row * kH + lr] = g * m;
  }
}

// ---------------------------------------------------------------------------
// Kernel: QKV projections. mode 0: phi(rope(xWq^T)) -> bf16
//                          mode 1: phi(rope(xWk^T)) * gate -> bf16
//                          mode 2: xWv^T -> bf16
// Wave computes a 16x64 strip (one full head width) so RoPE pairing
// (n <-> n+32) stays in-register between accumulator tiles (0,2) and (1,3).
// ---------------------------------------------------------------------------
__global__ __launch_bounds__(256)
void proj_kernel(const float* __restrict__ X, const float* __restrict__ W,
                 const float* __restrict__ gate, __bf16* __restrict__ dst, int mode) {
  const int tid = threadIdx.x;
  const int L = tid & 31, wv = tid >> 5;
  const int lr = L & 15, lanehi = (L >> 4) & 1;
  const int r0 = blockIdx.x * 128 + wv * 16;
  const int head = blockIdx.y;
  const int nbase = head * kDH;

  v8f acc[4];
#pragma unroll
  for (int t = 0; t < 4; ++t) acc[t] = zero8();

  const float* arow = X + (size_t)(r0 + lr) * kDM;
  for (int k0 = 0; k0 < kDM; k0 += 32) {
    __builtin_prefetch(arow + k0 + 256, 0, 0);
    v16bf a = a_frag_f32(arow + k0, lanehi);
#pragma unroll
    for (int t = 0; t < 4; ++t) {
      const float* brow = W + (size_t)(nbase + t * 16 + lr) * kDM + k0;
      v16bf b = b_frag_f32(brow, lanehi);
      acc[t] = wmma_bf16(a, b, acc[t]);
    }
  }

  if (mode == 2) {
#pragma unroll
    for (int t = 0; t < 4; ++t)
#pragma unroll
      for (int i = 0; i < 8; ++i) {
        int row = r0 + i + 8 * lanehi;
        dst[(size_t)row * kDM + nbase + t * 16 + lr] = (__bf16)acc[t][i];
      }
    return;
  }

  // RoPE + phi (+ gate for K)
  const float LF = 9.2103403719761836f / 32.0f;        // ln(10000)/half
  const float invf_lo = __expf(-(float)lr * LF);       // freq index lr     (tiles 0,2)
  const float invf_hi = __expf(-(float)(16 + lr) * LF);// freq index 16+lr  (tiles 1,3)
#pragma unroll
  for (int i = 0; i < 8; ++i) {
    int row = r0 + i + 8 * lanehi;
    int s = row & (kS - 1);
    float g = 1.f;
    if (mode == 1) g = gate[(size_t)row * kH + head];  // gate already includes mask
    float f1 = (float)s * invf_lo, sn1 = __sinf(f1), cs1 = __cosf(f1);
    float f2 = (float)s * invf_hi, sn2 = __sinf(f2), cs2 = __cosf(f2);
    float x0 = acc[0][i], x2 = acc[2][i];
    float x1 = acc[1][i], x3 = acc[3][i];
    float o0 = phi_elu(x0 * cs1 - x2 * sn1) * g;
    float o2 = phi_elu(x0 * sn1 + x2 * cs1) * g;
    float o1 = phi_elu(x1 * cs2 - x3 * sn2) * g;
    float o3 = phi_elu(x1 * sn2 + x3 * cs2) * g;
    __bf16* orow = dst + (size_t)row * kDM + nbase;
    orow[lr]      = (__bf16)o0;
    orow[16 + lr] = (__bf16)o1;
    orow[32 + lr] = (__bf16)o2;
    orow[48 + lr] = (__bf16)o3;
  }
}

// ---------------------------------------------------------------------------
// Kernel: per chunk-head local state:  dM = K~^T V,  ds = colsum(K~)
// K~[j] = kg_j * r^(C-1-j).  Grid = B*H*NC blocks, 128 threads (4 waves).
// ---------------------------------------------------------------------------
__global__ __launch_bounds__(128)
void chunk_state_kernel(const __bf16* __restrict__ kg, const __bf16* __restrict__ v,
                        const float* __restrict__ theta,
                        float* __restrict__ Mbuf, float* __restrict__ sbuf) {
  __shared__ __bf16 ktil[kC * kDH];   // K~^T : [d][j]
  __shared__ __bf16 vtil[kC * kDH];   // V^T  : [e][j]
  const int tid = threadIdx.x;
  const int c = blockIdx.x % kNC;
  const int bh = blockIdx.x / kNC;
  const int h = bh % kH, b = bh / kH;
  const float logr = head_logr(theta, h);

  for (int idx = tid; idx < kC * kDH; idx += 128) {
    int j = idx >> 6, d = idx & 63;
    size_t off = ((size_t)b * kS + (size_t)c * kC + j) * kDM + (size_t)h * kDH + d;
    float wj = __expf((float)(kC - 1 - j) * logr);
    ktil[d * kC + j] = (__bf16)((float)kg[off] * wj);
    vtil[d * kC + j] = v[off];
  }
  __syncthreads();

  if (tid < kDH) {                       // ds[d] = sum_j K~[j][d]
    float s = 0.f;
    for (int j = 0; j < kC; ++j) s += (float)ktil[tid * kC + j];
    sbuf[((size_t)bh * kNC + c) * kDH + tid] = s;
  }

  const int L = tid & 31, wv = tid >> 5;
  const int lr = L & 15, lanehi = (L >> 4) & 1;
  const int dbase = wv * 16;
  v8f acc[4];
#pragma unroll
  for (int t = 0; t < 4; ++t) acc[t] = zero8();
#pragma unroll
  for (int k0 = 0; k0 < kC; k0 += 32) {
    v16bf a = a_frag_bf(ktil + (dbase + lr) * kC + k0, lanehi);
#pragma unroll
    for (int t = 0; t < 4; ++t) {
      v16bf bb = b_frag_bf(vtil + (t * 16 + lr) * kC + k0, lanehi);
      acc[t] = wmma_bf16(a, bb, acc[t]);
    }
  }
  float* Mout = Mbuf + ((size_t)bh * kNC + c) * (kDH * kDH);
#pragma unroll
  for (int t = 0; t < 4; ++t)
#pragma unroll
    for (int i = 0; i < 8; ++i) {
      int d = dbase + i + 8 * lanehi;
      int e = t * 16 + lr;
      Mout[d * kDH + e] = acc[t][i];
    }
}

// ---------------------------------------------------------------------------
// Kernel: exclusive scan over chunks:  buf[c] <- state_before_c,
//         state = r^C * state + dM_c    (fp32, bandwidth bound, 64 blocks)
// ---------------------------------------------------------------------------
__global__ __launch_bounds__(128)
void scan_kernel(const float* __restrict__ theta,
                 float* __restrict__ Mbuf, float* __restrict__ sbuf) {
  const int bh = blockIdx.x;
  const int h = bh % kH;
  const float logr = head_logr(theta, h);
  const float rC = __expf((float)kC * logr);
  const int tid = threadIdx.x;

  float st[32];
#pragma unroll
  for (int k = 0; k < 32; ++k) st[k] = 0.f;
  float* Mb = Mbuf + (size_t)bh * kNC * (kDH * kDH);
  for (int c = 0; c < kNC; ++c) {
    float* p = Mb + (size_t)c * (kDH * kDH);
#pragma unroll
    for (int k = 0; k < 32; ++k) {
      int idx = k * 128 + tid;
      float tmp = p[idx];
      p[idx] = st[k];
      st[k] = st[k] * rC + tmp;
    }
  }
  if (tid < kDH) {
    float s = 0.f;
    float* sb = sbuf + (size_t)bh * kNC * kDH;
    for (int c = 0; c < kNC; ++c) {
      float tmp = sb[c * kDH + tid];
      sb[c * kDH + tid] = s;
      s = s * rC + tmp;
    }
  }
}

// ---------------------------------------------------------------------------
// Kernel: per chunk-head output:
//   A   = QK^T (masked, * r^(t-j))
//   Y   = (A V + r^(t+1) * Q M_prev) / (rowsum(A) + r^(t+1) q.s_prev + eps) * mask
// ---------------------------------------------------------------------------
__global__ __launch_bounds__(128)
void chunk_out_kernel(const __bf16* __restrict__ phq, const __bf16* __restrict__ kg,
                      const __bf16* __restrict__ v, const float* __restrict__ theta,
                      const float* __restrict__ Mbuf, const float* __restrict__ sbuf,
                      const unsigned char* __restrict__ mask, __bf16* __restrict__ y) {
  __shared__ __bf16 qt[kC * kDH];     // [t][d]
  __shared__ __bf16 kgr[kC * kDH];    // [j][d]
  __shared__ __bf16 vt[kC * kDH];     // V^T [e][j]
  __shared__ float  mt[kDH * kDH];    // M_prev^T [e][d] (fp32 for accuracy)
  __shared__ __bf16 atile[kC * kC];   // [t][j]
  __shared__ float  sprev[kDH], rowsum[kC], qs[kC];

  const int tid = threadIdx.x;
  const int c = blockIdx.x % kNC;
  const int bh = blockIdx.x / kNC;
  const int h = bh % kH, b = bh / kH;
  const float logr = head_logr(theta, h);
  const float* Mp = Mbuf + ((size_t)bh * kNC + c) * (kDH * kDH);

  for (int idx = tid; idx < kC * kDH; idx += 128) {
    int t = idx >> 6, d = idx & 63;
    size_t off = ((size_t)b * kS + (size_t)c * kC + t) * kDM + (size_t)h * kDH + d;
    qt[t * kDH + d] = phq[off];
    kgr[t * kDH + d] = kg[off];
    vt[d * kC + t] = v[off];
    mt[(idx & 63) * kDH + (idx >> 6)] = Mp[idx];   // transpose M_prev -> [e][d]
  }
  if (tid < kDH) sprev[tid] = sbuf[((size_t)bh * kNC + c) * kDH + tid];
  __syncthreads();

  const int L = tid & 31, wv = tid >> 5;
  const int lr = L & 15, lanehi = (L >> 4) & 1;
  const int tbase = wv * 16;

  // GEMM2: raw attention scores A = Q Kg^T
  v8f a2[4];
#pragma unroll
  for (int t = 0; t < 4; ++t) a2[t] = zero8();
#pragma unroll
  for (int k0 = 0; k0 < kDH; k0 += 32) {
    v16bf a = a_frag_bf(qt + (tbase + lr) * kDH + k0, lanehi);
#pragma unroll
    for (int t = 0; t < 4; ++t) {
      v16bf bb = b_frag_bf(kgr + (t * 16 + lr) * kDH + k0, lanehi);
      a2[t] = wmma_bf16(a, bb, a2[t]);
    }
  }
  // causal mask + per-element decay r^(t-j) -> LDS (bf16)
#pragma unroll
  for (int t = 0; t < 4; ++t)
#pragma unroll
    for (int i = 0; i < 8; ++i) {
      int tl = tbase + i + 8 * lanehi;
      int jl = t * 16 + lr;
      float av = (jl <= tl) ? a2[t][i] * __expf((float)(tl - jl) * logr) : 0.f;
      atile[tl * kC + jl] = (__bf16)av;
    }
  __syncthreads();

  if (tid < kC) {                       // denominator pieces
    float rs = 0.f;
    for (int j = 0; j < kC; ++j) rs += (float)atile[tid * kC + j];
    rowsum[tid] = rs;
    float qd = 0.f;
    for (int d = 0; d < kDH; ++d) qd += (float)qt[tid * kDH + d] * sprev[d];
    qs[tid] = qd;
  }

  // GEMM1: Y_inter = Q M_prev     GEMM3: Y_intra = A V
  v8f a1[4], a3[4];
#pragma unroll
  for (int t = 0; t < 4; ++t) { a1[t] = zero8(); a3[t] = zero8(); }
#pragma unroll
  for (int k0 = 0; k0 < kDH; k0 += 32) {
    v16bf aq = a_frag_bf(qt + (tbase + lr) * kDH + k0, lanehi);
#pragma unroll
    for (int t = 0; t < 4; ++t) {
      v16bf bm = b_frag_f32(mt + (t * 16 + lr) * kDH + k0, lanehi);
      a1[t] = wmma_bf16(aq, bm, a1[t]);
    }
  }
#pragma unroll
  for (int k0 = 0; k0 < kC; k0 += 32) {
    v16bf aa = a_frag_bf(atile + (tbase + lr) * kC + k0, lanehi);
#pragma unroll
    for (int t = 0; t < 4; ++t) {
      v16bf bv = b_frag_bf(vt + (t * 16 + lr) * kC + k0, lanehi);
      a3[t] = wmma_bf16(aa, bv, a3[t]);
    }
  }
  __syncthreads();

#pragma unroll
  for (int t = 0; t < 4; ++t)
#pragma unroll
    for (int i = 0; i < 8; ++i) {
      int tl = tbase + i + 8 * lanehi;
      int e = t * 16 + lr;
      float rp1 = __expf((float)(tl + 1) * logr);
      float rden = fast_rcp(rowsum[tl] + rp1 * qs[tl] + 1e-6f);  // v_rcp_f32
      size_t grow = (size_t)b * kS + (size_t)c * kC + tl;
      float m = mask[grow] ? 1.f : 0.f;
      float yv = (a3[t][i] + rp1 * a1[t][i]) * rden * m;
      y[grow * kDM + (size_t)h * kDH + e] = (__bf16)yv;
    }
}

// ---------------------------------------------------------------------------
// Kernel: h = x + y @ Wo^T + bo   (WMMA, bf16 A from y, bf16-cvt B from Wo)
// ---------------------------------------------------------------------------
__global__ __launch_bounds__(256)
void out_proj_kernel(const __bf16* __restrict__ Y, const float* __restrict__ Wo,
                     const float* __restrict__ bo, const float* __restrict__ X,
                     float* __restrict__ out) {
  const int tid = threadIdx.x;
  const int L = tid & 31, wv = tid >> 5;
  const int lr = L & 15, lanehi = (L >> 4) & 1;
  const int r0 = blockIdx.x * 128 + wv * 16;
  const int nbase = blockIdx.y * kDH;

  v8f acc[4];
#pragma unroll
  for (int t = 0; t < 4; ++t) acc[t] = zero8();
  const __bf16* arow = Y + (size_t)(r0 + lr) * kDM;
  for (int k0 = 0; k0 < kDM; k0 += 32) {
    v16bf a = a_frag_bf(arow + k0, lanehi);
#pragma unroll
    for (int t = 0; t < 4; ++t) {
      const float* brow = Wo + (size_t)(nbase + t * 16 + lr) * kDM + k0;
      v16bf b = b_frag_f32(brow, lanehi);
      acc[t] = wmma_bf16(a, b, acc[t]);
    }
  }
#pragma unroll
  for (int t = 0; t < 4; ++t)
#pragma unroll
    for (int i = 0; i < 8; ++i) {
      int row = r0 + i + 8 * lanehi;
      int col = nbase + t * 16 + lr;
      out[(size_t)row * kDM + col] =
          acc[t][i] + bo[col] + X[(size_t)row * kDM + col];
    }
}

// ---------------------------------------------------------------------------
// Kernel: row LayerNorm in-place on d_out (one block per row)
// ---------------------------------------------------------------------------
__global__ __launch_bounds__(256)
void ln_kernel(float* __restrict__ out, const float* __restrict__ gamma,
               const float* __restrict__ beta) {
  __shared__ float red[256], red2[256];
  __shared__ float mu_s, rstd_s;
  const int tid = threadIdx.x;
  float* p = out + (size_t)blockIdx.x * kDM;
  float hv[4], s = 0.f, s2 = 0.f;
#pragma unroll
  for (int k = 0; k < 4; ++k) {
    hv[k] = p[tid + k * 256];
    s += hv[k];
    s2 += hv[k] * hv[k];
  }
  red[tid] = s; red2[tid] = s2;
  __syncthreads();
  for (int off = 128; off > 0; off >>= 1) {
    if (tid < off) { red[tid] += red[tid + off]; red2[tid] += red2[tid + off]; }
    __syncthreads();
  }
  if (tid == 0) {
    float mu = red[0] * (1.f / kDM);
    float var = red2[0] * (1.f / kDM) - mu * mu;
    mu_s = mu;
    rstd_s = rsqrtf(var + 1e-5f);
  }
  __syncthreads();
#pragma unroll
  for (int k = 0; k < 4; ++k) {
    int ci = tid + k * 256;
    p[ci] = (hv[k] - mu_s) * rstd_s * gamma[ci] + beta[ci];
  }
}

// ---------------------------------------------------------------------------
extern "C" void kernel_launch(void* const* d_in, const int* in_sizes, int n_in,
                              void* d_out, int out_size, void* d_ws, size_t ws_size,
                              hipStream_t stream) {
  (void)in_sizes; (void)n_in; (void)out_size; (void)ws_size;
  const float* x    = (const float*)d_in[0];
  const unsigned char* mask = (const unsigned char*)d_in[1];  // jax bool = u8
  const float* Wq   = (const float*)d_in[2];
  const float* Wk   = (const float*)d_in[3];
  const float* Wv   = (const float*)d_in[4];
  const float* Wo   = (const float*)d_in[5];
  const float* bo   = (const float*)d_in[6];
  const float* Wg   = (const float*)d_in[7];
  const float* bg   = (const float*)d_in[8];
  const float* th   = (const float*)d_in[9];
  const float* lng  = (const float*)d_in[10];
  const float* lnb  = (const float*)d_in[11];
  float* out        = (float*)d_out;

  // workspace layout (~194 MB)
  char* w = (char*)d_ws;
  const size_t szbf = (size_t)kROWS * kDM * sizeof(unsigned short);  // 32 MB
  __bf16* phq  = (__bf16*)w;  w += szbf;
  __bf16* kgb  = (__bf16*)w;  w += szbf;
  __bf16* vb   = (__bf16*)w;  w += szbf;
  __bf16* yb   = (__bf16*)w;  w += szbf;
  float*  gate = (float*)w;   w += (size_t)kROWS * kH * sizeof(float);           // 1 MB
  float*  Mbuf = (float*)w;   w += (size_t)kB * kH * kNC * kDH * kDH * sizeof(float); // 64 MB
  float*  sbuf = (float*)w;

  dim3 gProj(kROWS / 128, kH);

  gate_kernel<<<kROWS / 16, 32, 0, stream>>>(x, Wg, bg, mask, gate);
  proj_kernel<<<gProj, 256, 0, stream>>>(x, Wq, gate, phq, 0);
  proj_kernel<<<gProj, 256, 0, stream>>>(x, Wk, gate, kgb, 1);
  proj_kernel<<<gProj, 256, 0, stream>>>(x, Wv, gate, vb, 2);
  chunk_state_kernel<<<kB * kH * kNC, 128, 0, stream>>>(kgb, vb, th, Mbuf, sbuf);
  scan_kernel<<<kB * kH, 128, 0, stream>>>(th, Mbuf, sbuf);
  chunk_out_kernel<<<kB * kH * kNC, 128, 0, stream>>>(phq, kgb, vb, th, Mbuf, sbuf,
                                                      mask, yb);
  out_proj_kernel<<<gProj, 256, 0, stream>>>(yb, Wo, bo, x, out);
  ln_kernel<<<kROWS, 256, 0, stream>>>(out, lng, lnb);
}
